// Layer_87488483820059
// MI455X (gfx1250) — compile-verified, matrix operand order checked by reading, not run on hardware
//
#include <hip/hip_runtime.h>
#include <hip/hip_bf16.h>
#include <math.h>

typedef __attribute__((ext_vector_type(16))) __bf16 v16bf;
typedef __attribute__((ext_vector_type(8)))  float  v8f;
typedef __attribute__((ext_vector_type(4)))  unsigned u32x4;
typedef __attribute__((ext_vector_type(8)))  int      i32x8;
typedef __attribute__((ext_vector_type(4)))  int      i32x4;

#define BM 128
#define BN 128
#define BK 32
#define D_H 1024
#define P_LEN 4096
#define BS_N 4
#define FF_N 4096
#define MROWS (BS_N * P_LEN)   // 16384
#define MBLK  (MROWS / BM)     // 128 m-blocks
#define SM    16               // m-blocks per L2 supertile (2048 rows)
#define NC 64                  // scan chunks
#define CL 64                  // chunk length (NC*CL == P_LEN)

// ---------- helpers ----------

// float -> bf16 bits, round-to-nearest-even (no scalar __bf16 arithmetic)
__device__ __forceinline__ unsigned short f2bf(float f) {
  unsigned u = __float_as_uint(f);
  unsigned r = u + 0x7FFFu + ((u >> 16) & 1u);
  return (unsigned short)(r >> 16);
}

__device__ __forceinline__ v8f wmma_bf16(v16bf a, v16bf b, v8f c) {
  return __builtin_amdgcn_wmma_f32_16x16x32_bf16(false, a, false, b, (short)0, c,
                                                 false, false);
}

// L2 supertile swizzle: SM m-blocks iterate across ALL n-blocks before the
// next m-supertile, so activation slice + weight panel stay in the 192MB L2.
__device__ __forceinline__ void block_map(int bid, int nB, int& mb, int& nb) {
  const int per = SM * nB;
  const int sup = bid / per;
  const int rem = bid - sup * per;
  nb = rem / SM;
  mb = sup * SM + (rem - nb * SM);
}

// Branch-free concat-segment select (s_cselect chain, uniform).
__device__ __forceinline__ const float* seg_src(const float* s0, const float* s1,
                                                const float* s2, const float* s3, int k0) {
  const int s = k0 >> 10;
  return s == 0 ? s0 : (s == 1 ? s1 : (s == 2 ? s2 : s3));
}

// Load one 16x32 bf16 operand fragment from an LDS tile (row-major, ldk=BK).
// ISA layout: lanes 0-15 -> row, K {0..7,16..23}; lanes 16-31 -> K {8..15,24..31}
__device__ __forceinline__ v16bf ld_frag(const unsigned short* tile, int row0, int lane) {
  const int r  = lane & 15;
  const int kh = (lane >> 4) << 3;   // 0 or 8
  const unsigned short* p = tile + (row0 + r) * BK + kh;
  union { v16bf v; uint4 q[2]; } u;
  u.q[0] = *(const uint4*)(p);        // K kh+0..7
  u.q[1] = *(const uint4*)(p + 16);   // K kh+16..23
  return u.v;
}

// Wave tile 32x64: 2x4 grid of 16x16 accumulators, 8 WMMAs per K-step.
__device__ __forceinline__ void mma_step(const unsigned short* As, const unsigned short* Bs,
                                         int mRow, int nCol, int lane, v8f acc[2][4]) {
  v16bf a0 = ld_frag(As, mRow,      lane);
  v16bf a1 = ld_frag(As, mRow + 16, lane);
  v16bf b[4];
  #pragma unroll
  for (int j = 0; j < 4; ++j) b[j] = ld_frag(Bs, nCol + 16 * j, lane);
  #pragma unroll
  for (int j = 0; j < 4; ++j) {
    acc[0][j] = wmma_bf16(a0, b[j], acc[0][j]);
    acc[1][j] = wmma_bf16(a1, b[j], acc[1][j]);
  }
}

// Stage a ROWSxBK fp32 panel into LDS as bf16 bits.
template <int ROWS>
__device__ __forceinline__ void stage_f32(unsigned short* dst, const float* __restrict__ src,
                                          size_t ld, int r0, int k0, int tid) {
  for (int e = tid; e < ROWS * BK; e += 256) {
    int r = e >> 5, kc = e & 31;
    dst[e] = f2bf(src[(size_t)(r0 + r) * ld + (size_t)(k0 + kc)]);
  }
}

// Async global->LDS staging of a BMxBK bf16 panel (8KB), ASYNCcnt-tracked.
__device__ __forceinline__ void stage_bf16_async(unsigned short* dst,
                                                 const unsigned short* __restrict__ src,
                                                 size_t ld, int r0, int k0, int tid) {
  const unsigned ldsBase = (unsigned)(uintptr_t)dst;   // low 32 bits == LDS offset
  #pragma unroll
  for (int i = 0; i < (BM * BK) / (256 * 4); ++i) {    // 4 qwords per thread
    int e = tid + i * 256;
    int r = e >> 3, c = (e & 7) * 4;
    unsigned lds = ldsBase + (unsigned)e * 8u;
    unsigned long long g =
        (unsigned long long)(uintptr_t)(src + (size_t)(r0 + r) * ld + (size_t)(k0 + c));
    asm volatile("global_load_async_to_lds_b64 %0, %1, off"
                 :: "v"(lds), "v"(g) : "memory");
  }
}

__device__ __forceinline__ void wait_async0() {
#if __has_builtin(__builtin_amdgcn_s_wait_asynccnt)
  __builtin_amdgcn_s_wait_asynccnt(0);
#else
  asm volatile("s_wait_asynccnt 0" ::: "memory");
#endif
}

__device__ __forceinline__ void wait_tensor0() {
#if __has_builtin(__builtin_amdgcn_s_wait_tensorcnt)
  __builtin_amdgcn_s_wait_tensorcnt((short)0);
#else
  asm volatile("s_wait_tensorcnt 0" ::: "memory");
#endif
}

// Tensor Data Mover: one 2D-tile DMA (BM rows x BK bf16 cols) global->LDS.
// D# per CDNA5 ISA. Issued by a single wave.
// This toolchain exposes the 6-arg builtin: (g0, g1, g2, g3, g4, cpol).
__device__ __forceinline__ void stage_bf16_tdm(unsigned short* dstLds,
                                               const unsigned short* __restrict__ src,
                                               unsigned ldElems, int r0, int k0, int tid) {
#if __has_builtin(__builtin_amdgcn_tensor_load_to_lds)
  if ((tid >> 5) == 0) {   // wave 0 only: TDM ignores EXEC, one DMA per tile
    unsigned long long ga =
        (unsigned long long)(uintptr_t)(src + (size_t)r0 * ldElems + (size_t)k0);
    unsigned lds = (unsigned)(uintptr_t)dstLds;
    u32x4 g0;
    g0.x = 1u;                                              // count=1 (valid), no gather
    g0.y = lds;                                             // lds_addr
    g0.z = (unsigned)(ga & 0xFFFFFFFFu);                    // global_addr[31:0]
    g0.w = (unsigned)((ga >> 32) & 0x01FFFFFFu) | (2u << 30); // addr[56:32] | type=2
    const unsigned td0 = 1u << 20, td1 = 1u << 20;          // large tensor dims: no OOB clip
    i32x8 g1;
    g1[0] = (int)(1u << 16);                                // data_size=1 (2 bytes)
    g1[1] = (int)((td0 & 0xFFFFu) << 16);                   // tensor_dim0 lo
    g1[2] = (int)((td0 >> 16) & 0xFFFFu) | (int)((td1 & 0xFFFFu) << 16);
    g1[3] = (int)((td1 >> 16) & 0xFFFFu) | (int)((unsigned)BK << 16);  // tile_dim0=32
    g1[4] = (int)(unsigned)BM;                              // tile_dim1=128, tile_dim2=0
    g1[5] = (int)ldElems;                                   // tensor_dim0_stride[31:0]
    g1[6] = 0;
    g1[7] = 0;
    i32x4 gz4 = {0, 0, 0, 0};
    i32x8 gz8 = {0, 0, 0, 0, 0, 0, 0, 0};
    __builtin_amdgcn_tensor_load_to_lds(g0, g1, gz4, gz4, gz8, 0);
  }
#else
  stage_bf16_async(dstLds, src, ldElems, r0, k0, tid);
#endif
}

__device__ __forceinline__ float gelu_tanh(float x) {
  float x3 = x * x * x;
  float t  = tanhf(0.7978845608028654f * (x + 0.044715f * x3));
  return 0.5f * x * (1.0f + t);
}

__device__ __forceinline__ float block_sum(float v, float* red) {
  __syncthreads();
  #pragma unroll
  for (int o = 16; o > 0; o >>= 1) v += __shfl_down(v, o, 32);
  const int lane = threadIdx.x & 31, w = threadIdx.x >> 5;
  if (lane == 0) red[w] = v;
  __syncthreads();
  if (threadIdx.x == 0) {
    float t = 0.f;
    #pragma unroll
    for (int i = 0; i < 8; ++i) t += red[i];
    red[0] = t;
  }
  __syncthreads();
  return red[0];
}

// ---------- kernel 1: fused gate GEMM  ab = u @ W_ab^T (+rank-1 surprise) ----------
__global__ __launch_bounds__(256) void k_gate(
    const float* __restrict__ x,   const float* __restrict__ ypm,
    const float* __restrict__ ywm, const float* __restrict__ yem,
    const float* __restrict__ Wab, const float* __restrict__ bab,
    const float* __restrict__ sur, const float* __restrict__ carry,
    float* __restrict__ Aout, float* __restrict__ Bout)
{
  __shared__ __align__(16) unsigned short As[2][BM * BK];
  __shared__ __align__(16) unsigned short Bs[2][BN * BK];
  const int tid  = threadIdx.x;
  const int lane = tid & 31;
  const int wave = tid >> 5;
  int mb, nb;
  block_map(blockIdx.x, 2048 / BN, mb, nb);   // 16 n-blocks
  const int m0 = mb * BM;
  const int n0 = nb * BN;
  const int mRow = (wave >> 1) * 32;
  const int nCol = (wave & 1) * 64;
  v8f acc[2][4] = {};

  // prologue: stage k0 = 0 into buffer 0
  stage_f32<BM>(As[0], x, D_H, m0, 0, tid);
  stage_f32<BN>(Bs[0], Wab, 4097, n0, 0, tid);

  int cur = 0;
  for (int k0 = 0; k0 < 4 * D_H; k0 += BK, cur ^= 1) {
    __syncthreads();                         // staged buffer `cur` visible
    const int k1 = k0 + BK;
    if (k1 < 4 * D_H) {                      // stage next while computing
      const float* s1 = seg_src(x, ypm, ywm, yem, k1);
      stage_f32<BM>(As[cur ^ 1], s1, D_H, m0, k1 & (D_H - 1), tid);
      stage_f32<BN>(Bs[cur ^ 1], Wab, 4097, n0, k1, tid);
    }
    mma_step(As[cur], Bs[cur], mRow, nCol, lane, acc);
  }

  const bool isA = (n0 < D_H);               // block-uniform a/b half
  #pragma unroll
  for (int ti = 0; ti < 2; ++ti)
    #pragma unroll
    for (int tj = 0; tj < 4; ++tj)
      #pragma unroll
      for (int e = 0; e < 8; ++e) {
        int m = m0 + mRow + ti * 16 + e + ((lane >> 4) << 3);
        int n = n0 + nCol + tj * 16 + (lane & 15);
        float v = acc[ti][tj][e] + bab[n] + sur[m] * Wab[(size_t)n * 4097 + 4096];
        if (isA) {
          float a = 1.0f / (1.0f + __expf(-v));
          Aout[(size_t)m * D_H + n] = a * carry[m];         // a_eff
        } else {
          Bout[(size_t)m * D_H + (n - D_H)] = tanhf(v);     // b_gate
        }
      }
}

// ---------- scan: h_t = a_t h_{t-1} + b_t, 3-pass chunked associative scan ----------
__global__ __launch_bounds__(256) void k_scan1(const float* __restrict__ A,
                                               const float* __restrict__ B,
                                               float* __restrict__ aggA,
                                               float* __restrict__ aggB) {
  const int idx   = blockIdx.x * 256 + threadIdx.x;   // [0, BS*NC*D)
  const int d     = idx & (D_H - 1);
  const int chunk = (idx >> 10) & (NC - 1);
  const int b     = idx >> 16;
  size_t base = ((size_t)b * P_LEN + (size_t)chunk * CL) * D_H + d;
  float pa = 1.f, pb = 0.f;
  for (int i = 0; i < CL; ++i) {
    float a  = A[base + (size_t)i * D_H];
    float bb = B[base + (size_t)i * D_H];
    pa *= a;
    pb  = a * pb + bb;
  }
  aggA[idx] = pa;
  aggB[idx] = pb;
}

__global__ __launch_bounds__(256) void k_scan2(const float* __restrict__ h0,
                                               float* __restrict__ aggA,
                                               const float* __restrict__ aggB) {
  const int idx = blockIdx.x * 256 + threadIdx.x;     // [0, BS*D)
  const int d = idx & (D_H - 1);
  const int b = idx >> 10;
  float h = h0[idx];
  for (int c = 0; c < NC; ++c) {
    size_t j = ((size_t)b * NC + c) * D_H + d;
    float pa = aggA[j], pb = aggB[j];
    aggA[j] = h;               // overwrite with the prefix entering this chunk
    h = pa * h + pb;
  }
}

__global__ __launch_bounds__(256) void k_scan3(const float* __restrict__ A,
                                               const float* __restrict__ B,
                                               const float* __restrict__ aggA,
                                               float* __restrict__ H) {
  const int idx   = blockIdx.x * 256 + threadIdx.x;
  const int d     = idx & (D_H - 1);
  const int chunk = (idx >> 10) & (NC - 1);
  const int b     = idx >> 16;
  size_t base = ((size_t)b * P_LEN + (size_t)chunk * CL) * D_H + d;
  float h = aggA[idx];
  for (int i = 0; i < CL; ++i) {
    float a  = A[base + (size_t)i * D_H];
    float bb = B[base + (size_t)i * D_H];
    h = a * h + bb;
    H[base + (size_t)i * D_H] = h;
  }
}

// ---------- kernel: out_pre = h @ W_o^T + b_o + x ----------
__global__ __launch_bounds__(256) void k_outproj(
    const float* __restrict__ H, const float* __restrict__ Wo,
    const float* __restrict__ bo, const float* __restrict__ x,
    float* __restrict__ outpre)
{
  __shared__ __align__(16) unsigned short As[2][BM * BK];
  __shared__ __align__(16) unsigned short Bs[2][BN * BK];
  const int tid  = threadIdx.x;
  const int lane = tid & 31;
  const int wave = tid >> 5;
  int mb, nb;
  block_map(blockIdx.x, D_H / BN, mb, nb);    // 8 n-blocks
  const int m0 = mb * BM;
  const int n0 = nb * BN;
  const int mRow = (wave >> 1) * 32;
  const int nCol = (wave & 1) * 64;
  v8f acc[2][4] = {};

  stage_f32<BM>(As[0], H,  D_H, m0, 0, tid);
  stage_f32<BN>(Bs[0], Wo, D_H, n0, 0, tid);

  int cur = 0;
  for (int k0 = 0; k0 < D_H; k0 += BK, cur ^= 1) {
    __syncthreads();
    const int k1 = k0 + BK;
    if (k1 < D_H) {
      stage_f32<BM>(As[cur ^ 1], H,  D_H, m0, k1, tid);
      stage_f32<BN>(Bs[cur ^ 1], Wo, D_H, n0, k1, tid);
    }
    mma_step(As[cur], Bs[cur], mRow, nCol, lane, acc);
  }

  #pragma unroll
  for (int ti = 0; ti < 2; ++ti)
    #pragma unroll
    for (int tj = 0; tj < 4; ++tj)
      #pragma unroll
      for (int e = 0; e < 8; ++e) {
        int m = m0 + mRow + ti * 16 + e + ((lane >> 4) << 3);
        int n = n0 + nCol + tj * 16 + (lane & 15);
        outpre[(size_t)m * D_H + n] = acc[ti][tj][e] + bo[n] + x[(size_t)m * D_H + n];
      }
}

// ---------- kernel: double LayerNorm rows (out + ffn_in in bf16) ----------
__global__ __launch_bounds__(256) void k_ln(
    const float* __restrict__ pre,
    const float* __restrict__ g1, const float* __restrict__ b1,
    const float* __restrict__ g2, const float* __restrict__ b2,
    float* __restrict__ out, unsigned short* __restrict__ ffn_in)
{
  __shared__ float red[8];
  const int m   = blockIdx.x;
  const int tid = threadIdx.x;
  const float* row = pre + (size_t)m * D_H;

  float v[4];
  float s = 0.f;
  #pragma unroll
  for (int i = 0; i < 4; ++i) { v[i] = row[tid + i * 256]; s += v[i]; }
  float mu = block_sum(s, red) * (1.0f / D_H);
  float s2 = 0.f;
  #pragma unroll
  for (int i = 0; i < 4; ++i) { float d = v[i] - mu; s2 += d * d; }
  float inv = rsqrtf(block_sum(s2, red) * (1.0f / D_H) + 1e-5f);

  float o[4];
  s = 0.f;
  #pragma unroll
  for (int i = 0; i < 4; ++i) {
    int n = tid + i * 256;
    o[i] = (v[i] - mu) * inv * g1[n] + b1[n];
    out[(size_t)m * D_H + n] = o[i];
    s += o[i];
  }
  float mu2 = block_sum(s, red) * (1.0f / D_H);
  s2 = 0.f;
  #pragma unroll
  for (int i = 0; i < 4; ++i) { float d = o[i] - mu2; s2 += d * d; }
  float inv2 = rsqrtf(block_sum(s2, red) * (1.0f / D_H) + 1e-5f);
  #pragma unroll
  for (int i = 0; i < 4; ++i) {
    int n = tid + i * 256;
    float f = (o[i] - mu2) * inv2 * g2[n] + b2[n];
    ffn_in[(size_t)m * D_H + n] = f2bf(f);
  }
}

// ---------- kernel: hmid = gelu(ffn_in @ W1^T + b1), bf16 out ----------
// A operand staged via gfx1250 async global->LDS (ASYNCcnt path).
__global__ __launch_bounds__(256) void k_ffn1(
    const unsigned short* __restrict__ ffn_in, const float* __restrict__ W1,
    const float* __restrict__ bias1, unsigned short* __restrict__ hmid)
{
  __shared__ __align__(16) unsigned short As[2][BM * BK];
  __shared__ __align__(16) unsigned short Bs[2][BN * BK];
  const int tid  = threadIdx.x;
  const int lane = tid & 31;
  const int wave = tid >> 5;
  int mb, nb;
  block_map(blockIdx.x, FF_N / BN, mb, nb);   // 32 n-blocks
  const int m0 = mb * BM;
  const int n0 = nb * BN;
  const int mRow = (wave >> 1) * 32;
  const int nCol = (wave & 1) * 64;
  v8f acc[2][4] = {};

  stage_bf16_async(As[0], ffn_in, D_H, m0, 0, tid);
  stage_f32<BN>(Bs[0], W1, D_H, n0, 0, tid);

  int cur = 0;
  for (int k0 = 0; k0 < D_H; k0 += BK, cur ^= 1) {
    wait_async0();                           // this wave's async tiles landed
    __syncthreads();                         // everyone's staging visible
    const int k1 = k0 + BK;
    if (k1 < D_H) {
      stage_bf16_async(As[cur ^ 1], ffn_in, D_H, m0, k1, tid);
      stage_f32<BN>(Bs[cur ^ 1], W1, D_H, n0, k1, tid);
    }
    mma_step(As[cur], Bs[cur], mRow, nCol, lane, acc);
  }

  #pragma unroll
  for (int ti = 0; ti < 2; ++ti)
    #pragma unroll
    for (int tj = 0; tj < 4; ++tj)
      #pragma unroll
      for (int e = 0; e < 8; ++e) {
        int m = m0 + mRow + ti * 16 + e + ((lane >> 4) << 3);
        int n = n0 + nCol + tj * 16 + (lane & 15);
        float g = gelu_tanh(acc[ti][tj][e] + bias1[n]);
        hmid[(size_t)m * FF_N + n] = f2bf(g);
      }
}

// ---------- kernel: final = out + hmid @ W2^T + b2 ----------
// A operand staged via the Tensor Data Mover (TENSORcnt path).
__global__ __launch_bounds__(256) void k_ffn2(
    const unsigned short* __restrict__ hmid, const float* __restrict__ W2,
    const float* __restrict__ bias2, const float* __restrict__ outres,
    float* __restrict__ dout)
{
  __shared__ __align__(16) unsigned short As[2][BM * BK];
  __shared__ __align__(16) unsigned short Bs[2][BN * BK];
  const int tid  = threadIdx.x;
  const int lane = tid & 31;
  const int wave = tid >> 5;
  int mb, nb;
  block_map(blockIdx.x, D_H / BN, mb, nb);    // 8 n-blocks
  const int m0 = mb * BM;
  const int n0 = nb * BN;
  const int mRow = (wave >> 1) * 32;
  const int nCol = (wave & 1) * 64;
  v8f acc[2][4] = {};

  stage_bf16_tdm(As[0], hmid, FF_N, m0, 0, tid);
  stage_f32<BN>(Bs[0], W2, FF_N, n0, 0, tid);

  int cur = 0;
  for (int k0 = 0; k0 < FF_N; k0 += BK, cur ^= 1) {
    wait_tensor0();                          // TDM tile landed (wave 0's counter)
    __syncthreads();
    const int k1 = k0 + BK;
    if (k1 < FF_N) {
      stage_bf16_tdm(As[cur ^ 1], hmid, FF_N, m0, k1, tid);
      stage_f32<BN>(Bs[cur ^ 1], W2, FF_N, n0, k1, tid);
    }
    mma_step(As[cur], Bs[cur], mRow, nCol, lane, acc);
  }

  #pragma unroll
  for (int ti = 0; ti < 2; ++ti)
    #pragma unroll
    for (int tj = 0; tj < 4; ++tj)
      #pragma unroll
      for (int e = 0; e < 8; ++e) {
        int m = m0 + mRow + ti * 16 + e + ((lane >> 4) << 3);
        int n = n0 + nCol + tj * 16 + (lane & 15);
        dout[(size_t)m * D_H + n] =
            outres[(size_t)m * D_H + n] + acc[ti][tj][e] + bias2[n];
      }
}

// ---------- launch ----------
extern "C" void kernel_launch(void* const* d_in, const int* in_sizes, int n_in,
                              void* d_out, int out_size, void* d_ws, size_t ws_size,
                              hipStream_t stream) {
  const float* x     = (const float*)d_in[0];
  const float* ypm   = (const float*)d_in[1];
  const float* ywm   = (const float*)d_in[2];
  const float* yem   = (const float*)d_in[3];
  const float* sur   = (const float*)d_in[4];
  const float* carry = (const float*)d_in[5];
  const float* h0    = (const float*)d_in[6];
  const float* Wab   = (const float*)d_in[7];
  const float* bab   = (const float*)d_in[8];
  const float* Wo    = (const float*)d_in[9];
  const float* bo    = (const float*)d_in[10];
  const float* lng   = (const float*)d_in[11];
  const float* lnb   = (const float*)d_in[12];
  const float* flng  = (const float*)d_in[13];
  const float* flnb  = (const float*)d_in[14];
  const float* W1    = (const float*)d_in[15];
  const float* b1    = (const float*)d_in[16];
  const float* W2    = (const float*)d_in[17];
  const float* b2    = (const float*)d_in[18];

  char* ws = (char*)d_ws;
  const size_t MB = 1024ull * 1024ull;
  // [0,64MB)    a_eff  -> reused as OUT_PRE
  // [64,128MB)  b_gate -> reused as OUT
  // [128,192MB) h      -> reused as FFN_IN (bf16, 32MB)
  // [192,320MB) HMID (bf16)
  // [320,322MB) scan chunk aggregates
  float*          A      = (float*)(ws + 0 * MB);
  float*          Bg     = (float*)(ws + 64 * MB);
  float*          H      = (float*)(ws + 128 * MB);
  unsigned short* HMID   = (unsigned short*)(ws + 192 * MB);
  float*          aggA   = (float*)(ws + 320 * MB);
  float*          aggB   = (float*)(ws + 321 * MB);
  float*          OUTPRE = A;
  float*          OUT    = Bg;
  unsigned short* FFN_IN = (unsigned short*)H;

  k_gate   <<<MBLK * (2048 / BN), 256, 0, stream>>>(x, ypm, ywm, yem, Wab, bab, sur, carry, A, Bg);
  k_scan1  <<<(BS_N * NC * D_H) / 256, 256, 0, stream>>>(A, Bg, aggA, aggB);
  k_scan2  <<<(BS_N * D_H) / 256,      256, 0, stream>>>(h0, aggA, aggB);
  k_scan3  <<<(BS_N * NC * D_H) / 256, 256, 0, stream>>>(A, Bg, aggA, H);
  k_outproj<<<MBLK * (D_H / BN),  256, 0, stream>>>(H, Wo, bo, x, OUTPRE);
  k_ln     <<<MROWS, 256, 0, stream>>>(OUTPRE, lng, lnb, flng, flnb, OUT, FFN_IN);
  k_ffn1   <<<MBLK * (FF_N / BN), 256, 0, stream>>>(FFN_IN, W1, b1, HMID);
  k_ffn2   <<<MBLK * (D_H / BN),  256, 0, stream>>>(HMID, W2, b2, OUT, (float*)d_out);
}
